// CrossModalAttention_56573309223864
// MI455X (gfx1250) — compile-verified, matrix operand order checked by reading, not run on hardware
//
#include <hip/hip_runtime.h>

// ---------------------------------------------------------------------------
// CDNA5 (gfx1250, wave32) fused multi-head cross-attention
//   q = query@Wq+bq ; k = key@Wk+bk ; v = value@Wv+bv   (bf16 WMMA GEMMs)
//   attn = softmax(mask(q k^T / 8))                     (flash-style, WMMA)
//   out  = (attn v) @ Wo + bo                           (bf16 WMMA GEMM, f32 out)
// All matmuls run on v_wmma_f32_16x16x32_bf16.
// Round 2: double-buffered LDS staging in the GEMM, 64-key KV blocks +
// pipelined K-fragment loads + double-buffered V staging in attention.
// ---------------------------------------------------------------------------

typedef __bf16 v8bf  __attribute__((ext_vector_type(8)));
typedef __bf16 v16bf __attribute__((ext_vector_type(16)));
typedef float  v8f   __attribute__((ext_vector_type(8)));
typedef float  v4f   __attribute__((ext_vector_type(4)));

#define CMA_B    4
#define CMA_LQ   1024
#define CMA_LKV  2048
#define CMA_H    16
#define CMA_HD   64
#define CMA_OD   1024
#define GP       40   // GEMM LDS pitch: 32+8 pad, 80B rows, 16B aligned, conflict-free
#define AP       72   // attn LDS pitch: 64+8 pad, 144B rows, 16B aligned, conflict-free

__device__ __forceinline__ v8f wmma_bf16(v16bf a, v16bf b, v8f c) {
  return __builtin_amdgcn_wmma_f32_16x16x32_bf16(
      /*neg_a=*/false, a, /*neg_b=*/false, b,
      /*c_mod=*/(short)0, c, /*reuse_a=*/false, /*reuse_b=*/false);
}

// A-style fragment (16x32 bf16, one row per lane, K split per ISA layout).
// Also used for B fragments by reading an N-major (transposed) buffer.
__device__ __forceinline__ v16bf load_frag(const __bf16* p, int rowStride, int lane) {
  const int r  = lane & 15;
  const int kb = (lane & 16) ? 8 : 0;          // lanes 16-31 take K {8..15,24..31}
  const __bf16* rp = p + (size_t)r * rowStride + kb;
  v8bf lo = *(const v8bf*)(rp);                 // K {kb .. kb+7}
  v8bf hi = *(const v8bf*)(rp + 16);            // K {kb+16 .. kb+23}
  return __builtin_shufflevector(lo, hi, 0,1,2,3,4,5,6,7,8,9,10,11,12,13,14,15);
}

__device__ __forceinline__ int   f2i(float f) { return __builtin_bit_cast(int, f); }
__device__ __forceinline__ float i2f(int   i) { return __builtin_bit_cast(float, i); }

// xor-reductions over 16-lane halves (columns of a C tile) via ds_swizzle
__device__ __forceinline__ float red_max16(float v) {
  v = fmaxf(v, i2f(__builtin_amdgcn_ds_swizzle(f2i(v), 0x041F)));  // xor 1
  v = fmaxf(v, i2f(__builtin_amdgcn_ds_swizzle(f2i(v), 0x081F)));  // xor 2
  v = fmaxf(v, i2f(__builtin_amdgcn_ds_swizzle(f2i(v), 0x101F)));  // xor 4
  v = fmaxf(v, i2f(__builtin_amdgcn_ds_swizzle(f2i(v), 0x201F)));  // xor 8
  return v;
}
__device__ __forceinline__ float red_sum16(float v) {
  v += i2f(__builtin_amdgcn_ds_swizzle(f2i(v), 0x041F));
  v += i2f(__builtin_amdgcn_ds_swizzle(f2i(v), 0x081F));
  v += i2f(__builtin_amdgcn_ds_swizzle(f2i(v), 0x101F));
  v += i2f(__builtin_amdgcn_ds_swizzle(f2i(v), 0x201F));
  return v;
}

// ---------------------------------------------------------------------------
// Tensor-core GEMM: C[M,N] = A[M,K] * B[K,N] + bias[N]
// 256 threads = 8 waves; block tile 128x128; wave tile 32x64 (2x4 WMMA tiles).
// Double-buffered LDS: global loads for tile i+1 overlap WMMAs on tile i.
// A staged row-major, B staged transposed (N-major), bf16, pitch 40.
// ---------------------------------------------------------------------------
template <bool A_IS_BF16, bool OUT_F32>
__global__ __launch_bounds__(256) void cma_gemm_kernel(
    const void* __restrict__ Asrc, const float* __restrict__ Bsrc,
    const float* __restrict__ bias, void* __restrict__ Cdst,
    int M, int N, int K)
{
  __shared__ __align__(16) __bf16 As[2][128 * GP];
  __shared__ __align__(16) __bf16 Bs[2][128 * GP];

  const int tid   = threadIdx.x;
  const int lane  = tid & 31;
  const int wave  = tid >> 5;
  const int waveM = wave & 3;   // 4 waves along M (32 rows each)
  const int waveN = wave >> 2;  // 2 waves along N (64 cols each)
  const int mBlk  = blockIdx.y * 128;
  const int nBlk  = blockIdx.x * 128;

  const int aRow = tid >> 1;           // 0..127
  const int aCol = (tid & 1) * 16;     // 0 / 16
  const int bRow = tid >> 3;           // 0..31 (K)
  const int bCol = (tid & 7) * 16;     // 0..112 (N)

  auto loadA = [&](int k0, v8bf& p0, v8bf& p1) {
    if (A_IS_BF16) {
      const __bf16* s = (const __bf16*)Asrc + (size_t)(mBlk + aRow) * K + k0 + aCol;
      p0 = *(const v8bf*)s;
      p1 = *(const v8bf*)(s + 8);
    } else {
      const float* s = (const float*)Asrc + (size_t)(mBlk + aRow) * K + k0 + aCol;
      v4f f0 = *(const v4f*)(s + 0);
      v4f f1 = *(const v4f*)(s + 4);
      v4f f2 = *(const v4f*)(s + 8);
      v4f f3 = *(const v4f*)(s + 12);
#pragma unroll
      for (int i = 0; i < 4; ++i) {
        p0[i] = (__bf16)f0[i]; p0[i + 4] = (__bf16)f1[i];
        p1[i] = (__bf16)f2[i]; p1[i + 4] = (__bf16)f3[i];
      }
    }
  };
  auto loadB = [&](int k0, v8bf& q0, v8bf& q1) {
    const float* s = Bsrc + (size_t)(k0 + bRow) * N + nBlk + bCol;
    v4f f0 = *(const v4f*)(s + 0);
    v4f f1 = *(const v4f*)(s + 4);
    v4f f2 = *(const v4f*)(s + 8);
    v4f f3 = *(const v4f*)(s + 12);
#pragma unroll
    for (int i = 0; i < 4; ++i) {
      q0[i] = (__bf16)f0[i]; q0[i + 4] = (__bf16)f1[i];
      q1[i] = (__bf16)f2[i]; q1[i + 4] = (__bf16)f3[i];
    }
  };
  auto storeA = [&](int buf, v8bf p0, v8bf p1) {
    *(v8bf*)&As[buf][aRow * GP + aCol]     = p0;
    *(v8bf*)&As[buf][aRow * GP + aCol + 8] = p1;
  };
  auto storeB = [&](int buf, v8bf q0, v8bf q1) {
#pragma unroll
    for (int i = 0; i < 8; ++i) {
      Bs[buf][(bCol + i)     * GP + bRow] = q0[i];
      Bs[buf][(bCol + i + 8) * GP + bRow] = q1[i];
    }
  };

  v8f acc[2][4] = {};

  // prologue: stage tile 0
  {
    v8bf a0, a1, b0, b1;
    loadA(0, a0, a1);
    loadB(0, b0, b1);
    storeA(0, a0, a1);
    storeB(0, b0, b1);
  }
  int cur = 0;
  for (int k0 = 0; k0 < K; k0 += 32) {
    __syncthreads();                       // Ls[cur] ready; prev reads done
    const bool hasNext = (k0 + 32) < K;
    v8bf na0, na1, nb0, nb1;
    if (hasNext) {                         // issue next tile's global loads early
      loadA(k0 + 32, na0, na1);
      loadB(k0 + 32, nb0, nb1);
    }
    // compute from buffer `cur`
    v16bf bfrag[4];
#pragma unroll
    for (int n = 0; n < 4; ++n)
      bfrag[n] = load_frag(&Bs[cur][(waveN * 64 + n * 16) * GP], GP, lane);
#pragma unroll
    for (int m = 0; m < 2; ++m) {
      v16bf afrag = load_frag(&As[cur][(waveM * 32 + m * 16) * GP], GP, lane);
#pragma unroll
      for (int n = 0; n < 4; ++n)
        acc[m][n] = wmma_bf16(afrag, bfrag[n], acc[m][n]);
    }
    if (hasNext) {
      storeA(cur ^ 1, na0, na1);
      storeB(cur ^ 1, nb0, nb1);
      cur ^= 1;
    }
  }

  // epilogue: bias + store (C layout: VGPR r -> row r + 8*(lane>=16), col lane&15)
  const int col = lane & 15;
  const int rh  = (lane >> 4) << 3;
#pragma unroll
  for (int m = 0; m < 2; ++m) {
#pragma unroll
    for (int n = 0; n < 4; ++n) {
      const int gcol = nBlk + waveN * 64 + n * 16 + col;
      const float bb = bias[gcol];
#pragma unroll
      for (int r = 0; r < 8; ++r) {
        const size_t grow = (size_t)(mBlk + waveM * 32 + m * 16 + rh + r);
        const float vv = acc[m][n][r] + bb;
        if (OUT_F32) ((float*)Cdst)[grow * N + gcol] = vv;
        else         ((__bf16*)Cdst)[grow * N + gcol] = (__bf16)vv;
      }
    }
  }
}

// ---------------------------------------------------------------------------
// Flash-style attention: one block per (b, h, 64 q-rows); 4 waves x 16 rows.
// 64-key KV blocks: per step 8 WMMAs for S = Q K^T (4 key tiles x 2 k-steps),
// one masked online-softmax pass (ds_swizzle reductions amortized over 64
// keys), P relayout through LDS, 8 WMMAs for O += P V. V double-buffered.
// ---------------------------------------------------------------------------
__global__ __launch_bounds__(128) void cma_attn_kernel(
    const __bf16* __restrict__ q_ws, const __bf16* __restrict__ k_ws,
    const __bf16* __restrict__ v_ws, const unsigned char* __restrict__ mask,
    __bf16* __restrict__ att_ws)
{
  __shared__ __align__(16) __bf16 Vt[2][CMA_HD * AP];   // [dim][key], 64 keys
  __shared__ __align__(16) __bf16 Ps[4][16 * AP];       // per-wave P (16 x 64)

  const int tid  = threadIdx.x;
  const int lane = tid & 31;
  const int wave = tid >> 5;
  const int b  = blockIdx.z;
  const int h  = blockIdx.y;
  const int qRow0 = blockIdx.x * 64 + wave * 16;        // in [0,1024)

  const size_t qbase = ((size_t)b * CMA_LQ + qRow0) * CMA_OD + h * CMA_HD;
  const v16bf qf0 = load_frag(q_ws + qbase,      CMA_OD, lane);  // dims 0..31
  const v16bf qf1 = load_frag(q_ws + qbase + 32, CMA_OD, lane);  // dims 32..63

  v8f o[4] = {};
  float mrow[8], lrow[8];
#pragma unroll
  for (int r = 0; r < 8; ++r) { mrow[r] = -1e30f; lrow[r] = 0.0f; }

  // V staging coords: 128 threads cover 64 keys x 64 dims (32 elems each)
  const int vKey  = tid & 63;
  const int vDim0 = (tid >> 6) * 32;

  auto loadV = [&](int kv, v8bf vr[4]) {
    const __bf16* s = v_ws + ((size_t)b * CMA_LKV + kv + vKey) * CMA_OD
                    + h * CMA_HD + vDim0;
#pragma unroll
    for (int j = 0; j < 4; ++j) vr[j] = *(const v8bf*)(s + j * 8);
  };
  auto storeV = [&](int buf, v8bf vr[4]) {
#pragma unroll
    for (int j = 0; j < 4; ++j)
#pragma unroll
      for (int i = 0; i < 8; ++i)
        Vt[buf][(vDim0 + j * 8 + i) * AP + vKey] = vr[j][i];
  };

  // prologue: stage V block 0
  {
    v8bf vr[4];
    loadV(0, vr);
    storeV(0, vr);
  }
  int vcur = 0;

  for (int kv = 0; kv < CMA_LKV; kv += 64) {
    __syncthreads();                      // Vt[vcur] ready; prev iter reads done
    const bool hasNext = (kv + 64) < CMA_LKV;
    v8bf vr[4];
    if (hasNext) loadV(kv + 64, vr);      // next V block in flight during S pass

    // ---- load all K fragments up front (partial waits overlap WMMAs) ----
    const __bf16* kbase = k_ws + ((size_t)b * CMA_LKV + kv) * CMA_OD + h * CMA_HD;
    if (hasNext)
      __builtin_prefetch(kbase + (size_t)64 * CMA_OD, 0, 1);
    v16bf kf[4][2];
#pragma unroll
    for (int t = 0; t < 4; ++t) {
      const __bf16* kb = kbase + (size_t)t * 16 * CMA_OD;   // key rows = B cols
      kf[t][0] = load_frag(kb,      CMA_OD, lane);
      kf[t][1] = load_frag(kb + 32, CMA_OD, lane);
    }

    // ---- S = (Q K^T) * scale + mask : four 16x16 tiles (64 keys) ----
    v8f s4[4];
#pragma unroll
    for (int t = 0; t < 4; ++t) {
      v8f z = {};
      z = wmma_bf16(qf0, kf[t][0], z);
      z = wmma_bf16(qf1, kf[t][1], z);
      s4[t] = z;
    }
    const float scale = 0.125f;                               // 1/sqrt(64)
#pragma unroll
    for (int t = 0; t < 4; ++t) {
      const unsigned char mk = mask[(size_t)b * CMA_LKV + kv + t * 16 + (lane & 15)];
      const float madd = mk ? -1e30f : 0.0f;
#pragma unroll
      for (int r = 0; r < 8; ++r) s4[t][r] = s4[t][r] * scale + madd;
    }

    // ---- online softmax (row-wise over the 64 keys) ----
#pragma unroll
    for (int r = 0; r < 8; ++r) {
      float mx = fmaxf(fmaxf(s4[0][r], s4[1][r]), fmaxf(s4[2][r], s4[3][r]));
      mx = red_max16(mx);
      const float nm   = fmaxf(mrow[r], mx);
      const float corr = __expf(mrow[r] - nm);
      const float e0 = __expf(s4[0][r] - nm);
      const float e1 = __expf(s4[1][r] - nm);
      const float e2 = __expf(s4[2][r] - nm);
      const float e3 = __expf(s4[3][r] - nm);
      lrow[r] = lrow[r] * corr + red_sum16((e0 + e1) + (e2 + e3));
      mrow[r] = nm;
#pragma unroll
      for (int nt = 0; nt < 4; ++nt) o[nt][r] *= corr;
      s4[0][r] = e0; s4[1][r] = e1; s4[2][r] = e2; s4[3][r] = e3;
    }

    // ---- P: C layout -> A layout via per-wave LDS tile (16 x 64) ----
    __bf16* pw = &Ps[wave][0];
    const int prh = (lane >> 4) << 3;
#pragma unroll
    for (int r = 0; r < 8; ++r) {
      const int row = r + prh;
#pragma unroll
      for (int t = 0; t < 4; ++t)
        pw[row * AP + t * 16 + (lane & 15)] = (__bf16)s4[t][r];
    }
    if (hasNext) storeV(vcur ^ 1, vr);    // stage next V into alternate buffer
    __syncthreads();                      // P + Vt[nxt] visible

    // ---- O += P (16x64) * V (64 x 64) : 4 n-tiles x 2 key-chunks ----
    const v16bf pf0 = load_frag(pw,      AP, lane);           // keys 0..31
    const v16bf pf1 = load_frag(pw + 32, AP, lane);           // keys 32..63
#pragma unroll
    for (int nt = 0; nt < 4; ++nt) {
      const __bf16* vtb = &Vt[vcur][(nt * 16) * AP];
      v16bf vf0 = load_frag(vtb,      AP, lane);
      v16bf vf1 = load_frag(vtb + 32, AP, lane);
      o[nt] = wmma_bf16(pf0, vf0, o[nt]);
      o[nt] = wmma_bf16(pf1, vf1, o[nt]);
    }
    if (hasNext) vcur ^= 1;
  }

  // ---- normalize and write attended (bf16) ----
  const int col = lane & 15;
  const int rh  = (lane >> 4) << 3;
#pragma unroll
  for (int r = 0; r < 8; ++r) {
    const float inv = 1.0f / lrow[r];
    const size_t row = (size_t)b * CMA_LQ + qRow0 + rh + r;
#pragma unroll
    for (int nt = 0; nt < 4; ++nt)
      att_ws[row * CMA_OD + h * CMA_HD + nt * 16 + col] = (__bf16)(o[nt][r] * inv);
  }
}

// ---------------------------------------------------------------------------
extern "C" void kernel_launch(void* const* d_in, const int* in_sizes, int n_in,
                              void* d_out, int out_size, void* d_ws, size_t ws_size,
                              hipStream_t stream) {
  (void)in_sizes; (void)n_in; (void)out_size; (void)ws_size;
  const float* query = (const float*)d_in[0];
  const float* key   = (const float*)d_in[1];
  const float* value = (const float*)d_in[2];
  const unsigned char* mask = (const unsigned char*)d_in[3];  // jnp bool -> 1 byte
  const float* Wq = (const float*)d_in[4];
  const float* bq = (const float*)d_in[5];
  const float* Wk = (const float*)d_in[6];
  const float* bk = (const float*)d_in[7];
  const float* Wv = (const float*)d_in[8];
  const float* bv = (const float*)d_in[9];
  const float* Wo = (const float*)d_in[10];
  const float* bo = (const float*)d_in[11];
  float* out = (float*)d_out;

  // workspace layout (bf16): q 8MB | k 16MB | v 16MB | attended 8MB = 48MB
  char* ws = (char*)d_ws;
  __bf16* q_ws = (__bf16*)(ws);
  __bf16* k_ws = (__bf16*)(ws + (size_t)8  * 1024 * 1024);
  __bf16* v_ws = (__bf16*)(ws + (size_t)24 * 1024 * 1024);
  __bf16* a_ws = (__bf16*)(ws + (size_t)40 * 1024 * 1024);

  const dim3 gblk(256);
  // q = query @ Wq + bq : [4096,1024] = [4096,1024]x[1024,1024]
  cma_gemm_kernel<false, false><<<dim3(8, 32), gblk, 0, stream>>>(
      (const void*)query, Wq, bq, (void*)q_ws, CMA_B * CMA_LQ, CMA_OD, 1024);
  // k = key @ Wk + bk : [8192,1024] = [8192,768]x[768,1024]
  cma_gemm_kernel<false, false><<<dim3(8, 64), gblk, 0, stream>>>(
      (const void*)key, Wk, bk, (void*)k_ws, CMA_B * CMA_LKV, CMA_OD, 768);
  // v = value @ Wv + bv
  cma_gemm_kernel<false, false><<<dim3(8, 64), gblk, 0, stream>>>(
      (const void*)value, Wv, bv, (void*)v_ws, CMA_B * CMA_LKV, CMA_OD, 768);
  // attention: grid = (LQ/64, H, B), 128 threads (4 waves)
  cma_attn_kernel<<<dim3(CMA_LQ / 64, CMA_H, CMA_B), dim3(128), 0, stream>>>(
      q_ws, k_ws, v_ws, mask, a_ws);
  // out = attended @ Wo + bo (f32 output)
  cma_gemm_kernel<true, true><<<dim3(8, 32), gblk, 0, stream>>>(
      (const void*)a_ws, Wo, bo, (void*)out, CMA_B * CMA_LQ, CMA_OD, 1024);
}
